// GCNResidualBlock_75505525064552
// MI455X (gfx1250) — matrix operand order repaired and checked by reading, not compiled, not run on hardware
//
#include <hip/hip_runtime.h>
#include <hip/hip_bf16.h>

typedef __attribute__((ext_vector_type(16))) _Float16 v16h;
typedef __attribute__((ext_vector_type(8)))  _Float16 v8h;
typedef __attribute__((ext_vector_type(8)))  float    v8f;

#define F 64
#define EPS 1e-5f

// ---------------------------------------------------------------------------
// zero-fill float buffer, float4 per thread
// ---------------------------------------------------------------------------
__global__ void zero_f32v4(float4* __restrict__ p, int n4) {
    int i = blockIdx.x * blockDim.x + threadIdx.x;
    if (i < n4) p[i] = make_float4(0.f, 0.f, 0.f, 0.f);
}

// ---------------------------------------------------------------------------
// deg[dst] += ew[e]   (self-loop +1 added in rsqrt kernel)
// ---------------------------------------------------------------------------
__global__ void degree_kernel(const long long* __restrict__ ei,
                              const float* __restrict__ ew,
                              float* __restrict__ deg, int E) {
    int e = blockIdx.x * blockDim.x + threadIdx.x;
    if (e < E) {
        long long d = ei[(long long)E + e];
        atomicAdd(&deg[d], ew[e]);
    }
}

// dinv[i] = rsqrt(deg[i] + 1)
__global__ void dinv_kernel(const float* __restrict__ deg,
                            float* __restrict__ dinv, int N) {
    int i = blockIdx.x * blockDim.x + threadIdx.x;
    if (i < N) dinv[i] = __frsqrt_rn(deg[i] + 1.0f);
}

// ---------------------------------------------------------------------------
// H = X @ W  (f32 in, f16 WMMA, f32 accumulate/out). X: N x 64, W: 64 x 64.
// Block = 256 threads = 8 waves; block computes 32 rows x 64 cols
// (2 row-tiles x 4 col-tiles of 16x16, one tile per wave, K=64 -> 2 WMMA).
// H is padded to a multiple of 32 rows -> unconditional coalesced writeback
// staged through LDS (two global_store_b128 per thread).
// ---------------------------------------------------------------------------
__global__ void gemm_wmma_kernel(const float* __restrict__ X,
                                 const float* __restrict__ W,
                                 float* __restrict__ H, int N) {
    __shared__ _Float16 sWT[F * F];   // sWT[n*64 + k] = W[k][n]   (8 KB)
    __shared__ _Float16 sX[32 * F];   // 32 rows of X in f16       (4 KB)
    __shared__ float    sC[32 * F];   // C staging for coalesced WB (8 KB)

    int tid = threadIdx.x;
    int rowBase = blockIdx.x * 32;

    // load + transpose W to f16
    for (int i = tid; i < F * F; i += 256) {
        int k = i >> 6, n = i & 63;
        sWT[n * F + k] = (_Float16)W[i];
    }
    // load 32 rows of X to f16 (pad OOB rows with 0)
    for (int i = tid; i < 32 * F; i += 256) {
        int gr = rowBase + (i >> 6);
        sX[i] = (gr < N) ? (_Float16)X[(long long)gr * F + (i & 63)] : (_Float16)0.0f;
    }
    __syncthreads();

    int wave = tid >> 5, lane = tid & 31;
    int rowTile = (wave >> 2) * 16;   // 0 or 16
    int colTile = (wave & 3) * 16;    // 0,16,32,48
    int m  = lane & 15;               // row (A) / col (B) within tile
    int kh = lane >> 4;               // lane-half selects K sub-block

    v8f c = {};
#pragma unroll
    for (int kk = 0; kk < F; kk += 32) {
        // A fragment (contiguous k order, same as B): lanes 0-15 K=0..15,
        // lanes 16-31 K=16..31 -> two contiguous 8-half chunks.
        v8h alo = *(const v8h*)&sX[(rowTile + m) * F + kk + kh * 16];
        v8h ahi = *(const v8h*)&sX[(rowTile + m) * F + kk + kh * 16 + 8];
        v16h a;
#pragma unroll
        for (int i = 0; i < 8; ++i) { a[i] = alo[i]; a[8 + i] = ahi[i]; }

        // B fragment (32x16): lane n = m, lanes 0-15 K=0..15, lanes 16-31
        // K=16..31; transposed-W storage makes the 16 K-values contiguous.
        v16h b = *(const v16h*)&sWT[(colTile + m) * F + kk + kh * 16];

        c = __builtin_amdgcn_wmma_f32_16x16x32_f16(
                /*neg_a=*/false, a, /*neg_b=*/false, b,
                /*c_mod=*/(short)0, c, /*reuse_a=*/false, /*reuse_b=*/false);
    }

    // C layout: VGPR v -> M = v + 8*(lane>=16), N = lane&15 -> stage in LDS
    int col = colTile + m;
#pragma unroll
    for (int v = 0; v < 8; ++v)
        sC[(rowTile + v + (kh << 3)) * F + col] = c[v];
    __syncthreads();

    // coalesced writeback: thread -> row tid/8, 8 consecutive cols.
    // H rows are padded to a multiple of 32, so no bounds check needed
    // (padded rows are exact zeros: padded A rows were zero-filled).
    int r = tid >> 3, cb = (tid & 7) * 8;
    float4* dst = (float4*)&H[(long long)(rowBase + r) * F + cb];
    const float4* srcv = (const float4*)&sC[r * F + cb];
    dst[0] = srcv[0];
    dst[1] = srcv[1];
}

// ---------------------------------------------------------------------------
// agg[dst] += dinv[src]*ew*dinv[dst] * h[src]
// One wave per edge, 2 features per lane, fp32 global atomics.
// ---------------------------------------------------------------------------
__global__ void edge_agg_kernel(const long long* __restrict__ ei,
                                const float* __restrict__ ew,
                                const float* __restrict__ dinv,
                                const float* __restrict__ h,
                                float* __restrict__ agg, int E) {
    int e = blockIdx.x * 8 + (threadIdx.x >> 5);
    int lane = threadIdx.x & 31;
    if (e >= E) return;
    // stream-prefetch the index arrays ahead
    if (lane == 0 && e + 4096 < E) {
        __builtin_prefetch(&ei[e + 4096], 0, 1);
        __builtin_prefetch(&ei[(long long)E + e + 4096], 0, 1);
    }
    long long s = ei[e];
    long long d = ei[(long long)E + e];
    float nrm = dinv[s] * ew[e] * dinv[d];
    int f = lane * 2;
    float2 hv = *(const float2*)(h + s * F + f);
    atomicAdd(&agg[d * F + f],     nrm * hv.x);
    atomicAdd(&agg[d * F + f + 1], nrm * hv.y);
}

// ---------------------------------------------------------------------------
// pre-BN value: v = agg + h * dinv^2 + bias; write in place, accumulate
// per-column sum / sumsq via LDS reduction + global atomics.
// Block = 256 threads = 4 rows x 64 cols.
// ---------------------------------------------------------------------------
__global__ void selfloop_stats_kernel(float* __restrict__ agg,
                                      const float* __restrict__ h,
                                      const float* __restrict__ dinv,
                                      const float* __restrict__ bias,
                                      float* __restrict__ stats, int N) {
    __shared__ float ssum[256];
    __shared__ float ssq[256];
    int col = threadIdx.x & 63;
    int r   = threadIdx.x >> 6;
    int row = blockIdx.x * 4 + r;
    float val = 0.0f;
    if (row < N) {
        float di = dinv[row];
        long long idx = (long long)row * F + col;
        val = agg[idx] + h[idx] * (di * di) + bias[col];
        agg[idx] = val;
    }
    ssum[threadIdx.x] = val;
    ssq[threadIdx.x]  = val * val;
    __syncthreads();
    if (r == 0) {
        float s = ssum[col] + ssum[64 + col] + ssum[128 + col] + ssum[192 + col];
        float q = ssq[col]  + ssq[64 + col]  + ssq[128 + col]  + ssq[192 + col];
        atomicAdd(&stats[col], s);
        atomicAdd(&stats[64 + col], q);
    }
}

// ---------------------------------------------------------------------------
// y = (v - mean) * rsqrt(var+eps) * gamma + beta  [+ residual] -> relu -> out
// float4 per thread (4 consecutive cols of one row).
// ---------------------------------------------------------------------------
__global__ void bn_apply_kernel(const float4* __restrict__ v,
                                const float* __restrict__ stats,
                                const float* __restrict__ gamma,
                                const float* __restrict__ beta,
                                const float4* __restrict__ residual,
                                float4* __restrict__ out, int N) {
    long long i = (long long)blockIdx.x * blockDim.x + threadIdx.x;  // vec4 idx
    long long total4 = (long long)N * (F / 4);
    if (i >= total4) return;
    int cb = (int)((i & 15) << 2);   // starting column of this float4
    float invN = 1.0f / (float)N;
    float4 val = v[i];
    float* vp = &val.x;
    float4 res;
    if (residual) res = residual[i];
    float* rp = &res.x;
    float4 o;
    float* op = &o.x;
#pragma unroll
    for (int j = 0; j < 4; ++j) {
        int col = cb + j;
        float mean = stats[col] * invN;
        float var  = stats[64 + col] * invN - mean * mean;
        float istd = __frsqrt_rn(var + EPS);
        float y = (vp[j] - mean) * istd * gamma[col] + beta[col];
        if (residual) y += rp[j];
        op[j] = fmaxf(y, 0.0f);
    }
    out[i] = o;
}

// ---------------------------------------------------------------------------
// launch
// ---------------------------------------------------------------------------
extern "C" void kernel_launch(void* const* d_in, const int* in_sizes, int n_in,
                              void* d_out, int out_size, void* d_ws, size_t ws_size,
                              hipStream_t stream) {
    const float*     x   = (const float*)d_in[0];
    const long long* ei  = (const long long*)d_in[1];   // int64 [2,E]
    const float*     ew  = (const float*)d_in[2];
    const float*     W1  = (const float*)d_in[3];
    const float*     b1  = (const float*)d_in[4];
    const float*     g1  = (const float*)d_in[5];
    const float*     be1 = (const float*)d_in[6];
    const float*     W2  = (const float*)d_in[7];
    const float*     b2  = (const float*)d_in[8];
    const float*     g2  = (const float*)d_in[9];
    const float*     be2 = (const float*)d_in[10];
    float*           out = (float*)d_out;

    const int N    = in_sizes[0] / F;       // 50000
    const int E    = in_sizes[2];           // 800000
    const int Npad = (N + 31) & ~31;        // GEMM output padded to 32 rows

    // workspace layout (floats)
    float* ws    = (float*)d_ws;
    float* deg   = ws;                          // N
    float* dinv  = deg + N;                     // N
    float* B0    = dinv + N;                    // Npad*F : h (gemm out, padded)
    float* B1    = B0 + (size_t)Npad * F;       // Npad*F : agg / pre-BN
    float* B2    = B1 + (size_t)Npad * F;       // Npad*F : act1
    float* stats = B2 + (size_t)Npad * F;       // 128    : col sum / sumsq

    const int NF = N * F;
    dim3 blk(256);
    dim3 gN((N + 255) / 256);
    dim3 gE((E + 255) / 256);
    dim3 gNF4((NF / 4 + 255) / 256);
    dim3 gEdge((E + 7) / 8);          // 8 edges per 256-thread block
    dim3 gGemm(Npad / 32);            // 32 rows per block
    dim3 gRows((N + 3) / 4);          // 4 rows per block

    // ---------------- degree / normalization ----------------
    zero_f32v4<<<dim3((N / 4 + 255) / 256), blk, 0, stream>>>((float4*)deg, N / 4);
    zero_f32v4<<<gNF4, blk, 0, stream>>>((float4*)B1, NF / 4);
    zero_f32v4<<<dim3(1), dim3(32), 0, stream>>>((float4*)stats, 32);
    degree_kernel<<<gE, blk, 0, stream>>>(ei, ew, deg, E);
    dinv_kernel<<<gN, blk, 0, stream>>>(deg, dinv, N);

    // ---------------- conv1 + bn1 + relu ----------------
    gemm_wmma_kernel<<<gGemm, blk, 0, stream>>>(x, W1, B0, N);
    edge_agg_kernel<<<gEdge, blk, 0, stream>>>(ei, ew, dinv, B0, B1, E);
    selfloop_stats_kernel<<<gRows, blk, 0, stream>>>(B1, B0, dinv, b1, stats, N);
    bn_apply_kernel<<<gNF4, blk, 0, stream>>>((const float4*)B1, stats, g1, be1,
                                              nullptr, (float4*)B2, N);

    // ---------------- conv2 + bn2 + residual + relu ----------------
    zero_f32v4<<<gNF4, blk, 0, stream>>>((float4*)B1, NF / 4);
    zero_f32v4<<<dim3(1), dim3(32), 0, stream>>>((float4*)stats, 32);
    gemm_wmma_kernel<<<gGemm, blk, 0, stream>>>(B2, W2, B0, N);
    edge_agg_kernel<<<gEdge, blk, 0, stream>>>(ei, ew, dinv, B0, B1, E);
    selfloop_stats_kernel<<<gRows, blk, 0, stream>>>(B1, B0, dinv, b2, stats, N);
    bn_apply_kernel<<<gNF4, blk, 0, stream>>>((const float4*)B1, stats, g2, be2,
                                              (const float4*)x, (float4*)out, N);
}
